// FlakyGAT_1657857376749
// MI455X (gfx1250) — matrix operand order repaired and checked by reading, not compiled, not run on hardware
//
#include <hip/hip_runtime.h>

// ---------------- types for CDNA5 WMMA ----------------
typedef __attribute__((ext_vector_type(16))) __bf16 v16bf;
typedef __attribute__((ext_vector_type(8)))  float  v8f;
typedef __attribute__((ext_vector_type(2)))  float  v2f;
typedef __attribute__((ext_vector_type(2)))  __bf16 v2bf;
typedef __attribute__((ext_vector_type(4)))  int    v4i;

union Frag { v16bf v; uint4 q[2]; };

#define HAS_ASYNC_LDS (__has_builtin(__builtin_amdgcn_global_load_async_to_lds_b128))

__device__ __forceinline__ unsigned short f2bf(float f) {
    unsigned u = __float_as_uint(f);
    unsigned r = u + 0x7fffu + ((u >> 16) & 1u);   // RNE truncate to bf16
    return (unsigned short)(r >> 16);
}

__device__ __forceinline__ unsigned pack_bf16x2(float x, float y) {
    v2f f; f.x = x; f.y = y;
    v2bf b = __builtin_convertvector(f, v2bf);     // -> v_cvt_pk_bf16_f32
    return __builtin_bit_cast(unsigned, b);
}

// order-preserving float<->uint encoding for atomicMax-based segment max
__device__ __forceinline__ unsigned enc_f(float f) {
    unsigned u = __float_as_uint(f);
    return (u & 0x80000000u) ? ~u : (u | 0x80000000u);
}
__device__ __forceinline__ float dec_f(unsigned u) {
    unsigned v = (u & 0x80000000u) ? (u & 0x7FFFFFFFu) : ~u;
    return __uint_as_float(v);
}
#define ENC_NEG_INF 0x007FFFFFu   // enc_f(-inf)

// ============================================================
// weight pre-transpose + bf16 convert: Wt[col][k] = bf16(W[k][col])
// ============================================================
__global__ __launch_bounds__(256)
void wt_convert_kernel(const float* __restrict__ W, unsigned short* __restrict__ Wt, int K) {
    int i = blockIdx.x * blockDim.x + threadIdx.x;
    if (i >= 64 * K) return;
    int col = i / K, k = i - col * K;
    Wt[(size_t)col * K + k] = f2bf(W[(size_t)k * 64 + col]);
}

// ============================================================
// GEMM  C[N x 64] = A[N x K] * Bt   (A fp32 -> bf16 in LDS, Bt pre-transposed bf16)
// block = 256 threads (8 waves), computes 128 rows x 64 cols
// B tile staged via GLOBAL_LOAD_ASYNC_TO_LDS_B128 (ASYNCcnt);
// A tile: 4 overlapped b128 loads -> packed cvt -> 2x ds_store_b128 per thread
// ============================================================
__global__ __launch_bounds__(256)
void gemm_bf16_wmma(const float* __restrict__ A, const unsigned short* __restrict__ Wt,
                    float* __restrict__ C, int N, int K) {
    __shared__ __align__(16) unsigned short As[128][40];  // 32 K + pad (80B row, 16B aligned)
    __shared__ __align__(16) unsigned short Bs[64][40];   // [col][K] bf16

    const int rowBase = blockIdx.x * 128;
    const int tid  = threadIdx.x;
    const int lane = tid & 31;
    const int wv   = tid >> 5;

    v8f acc[4] = {};

    // per-thread B-tile copy coordinates (16B per thread, 64 cols x 64B tile)
    const int bcol = tid >> 2;
    const int bseg = (tid & 3) * 8;               // ushort offset within the 32-K run

    // per-thread A-tile coordinates: one contiguous half-row (16 floats)
    const int ar = tid >> 1;                      // 0..127
    const int ak = (tid & 1) * 16;                // 0 or 16
    int agr = rowBase + ar; if (agr >= N) agr = N - 1;
    const float* arow = A + (size_t)agr * K + ak;

    for (int kc = 0; kc < K; kc += 32) {
        // ---- async-stage B tile: Wt[col][kc..kc+31] -> Bs[col][0..31] ----
        {
            const unsigned short* gsrc = Wt + (size_t)bcol * K + kc + bseg;
            unsigned short*       ldst = &Bs[bcol][bseg];
#if HAS_ASYNC_LDS
            __builtin_amdgcn_global_load_async_to_lds_b128(
                (__attribute__((address_space(1))) v4i*)gsrc,
                (__attribute__((address_space(3))) v4i*)ldst, 0, 0);
#else
            *(uint4*)ldst = *(const uint4*)gsrc;
#endif
        }
        // ---- stage A tile 128x32: issue all loads, then convert, then 2x b128 LDS store ----
        {
            const float4* ap = (const float4*)(arow + kc);
            float4 f0 = ap[0];
            float4 f1 = ap[1];
            float4 f2 = ap[2];
            float4 f3 = ap[3];
            uint4 q0, q1;
            q0.x = pack_bf16x2(f0.x, f0.y); q0.y = pack_bf16x2(f0.z, f0.w);
            q0.z = pack_bf16x2(f1.x, f1.y); q0.w = pack_bf16x2(f1.z, f1.w);
            q1.x = pack_bf16x2(f2.x, f2.y); q1.y = pack_bf16x2(f2.z, f2.w);
            q1.z = pack_bf16x2(f3.x, f3.y); q1.w = pack_bf16x2(f3.z, f3.w);
            *(uint4*)&As[ar][ak]     = q0;
            *(uint4*)&As[ar][ak + 8] = q1;
        }
        // prefetch next A chunk (global_prefetch_b8)
        if (kc + 32 < K) {
            __builtin_prefetch(arow + kc + 32, 0, 1);
        }
#if HAS_ASYNC_LDS
#if __has_builtin(__builtin_amdgcn_s_wait_asynccnt)
        __builtin_amdgcn_s_wait_asynccnt(0);
#else
        asm volatile("s_wait_asynccnt 0x0" ::: "memory");
#endif
#endif
        __syncthreads();

        // ---- A fragment: 16-bit A 16x32 layout (ISA 7.12.2) ----
        Frag a;
        const int rsel = 16 * wv + (lane & 15);
        const int kb   = (lane >> 4) * 8;     // lanes 0-15: K 0..7/16..23 ; 16-31: 8..15/24..31
        a.q[0] = *(const uint4*)&As[rsel][kb];
        a.q[1] = *(const uint4*)&As[rsel][kb + 16];

        #pragma unroll
        for (int t = 0; t < 4; ++t) {
            Frag b;
            const int col = 16 * t + (lane & 15);
            const int k0  = (lane >> 4) * 8;  // lanes 0-15: K 0..7/16..23 ; 16-31: 8..15/24..31
            b.q[0] = *(const uint4*)&Bs[col][k0];
            b.q[1] = *(const uint4*)&Bs[col][k0 + 16];
            acc[t] = __builtin_amdgcn_wmma_f32_16x16x32_bf16(
                         false, a.v, false, b.v, (short)0, acc[t], false, false);
        }
        __syncthreads();
    }

    // ---- store C per 32-bit C/D layout: VGPR r, lane l -> row 16w+r+8*(l>=16), col l&15 ----
    const int rhalf = (lane >> 4) * 8;
    const int n0    = lane & 15;
    #pragma unroll
    for (int t = 0; t < 4; ++t) {
        #pragma unroll
        for (int r = 0; r < 8; ++r) {
            int row = rowBase + 16 * wv + r + rhalf;
            if (row < N) C[(size_t)row * 64 + 16 * t + n0] = acc[t][r];
        }
    }
}

// ============================================================
// per-node attention dots: a_s = h . att_src, a_d = h . att_dst
// ============================================================
__global__ __launch_bounds__(256)
void attdot_kernel(const float* __restrict__ h, const float* __restrict__ asv,
                   const float* __restrict__ adv, float* __restrict__ a_s,
                   float* __restrict__ a_d, int N) {
    int gw   = (blockIdx.x * blockDim.x + threadIdx.x) >> 5;
    int lane = threadIdx.x & 31;
    int nw   = (gridDim.x * blockDim.x) >> 5;
    float sx = asv[2 * lane], sy = asv[2 * lane + 1];
    float dx = adv[2 * lane], dy = adv[2 * lane + 1];
    for (int node = gw; node < N; node += nw) {
        float2 hv = ((const float2*)(h + (size_t)node * 64))[lane];
        float s = hv.x * sx + hv.y * sy;
        float d = hv.x * dx + hv.y * dy;
        #pragma unroll
        for (int m = 16; m >= 1; m >>= 1) {
            s += __shfl_xor(s, m, 32);
            d += __shfl_xor(d, m, 32);
        }
        if (lane == 0) { a_s[node] = s; a_d[node] = d; }
    }
}

// ============================================================
// edge kernels (self-loops virtual: e >= E -> src=dst=e-E)
// ============================================================
__global__ __launch_bounds__(256)
void edge_max_kernel(const long long* __restrict__ ei, const float* __restrict__ a_s,
                     const float* __restrict__ a_d, unsigned* __restrict__ menc,
                     int E, int Etot) {
    int e = blockIdx.x * blockDim.x + threadIdx.x;
    if (e >= Etot) return;
    long long s, d;
    if (e < E) { s = ei[e]; d = ei[E + e]; } else { s = d = e - E; }
    float v = a_s[s] + a_d[d];
    v = v >= 0.f ? v : 0.2f * v;                    // LeakyReLU(0.2)
    atomicMax(&menc[(int)d], enc_f(v));
}

__global__ __launch_bounds__(256)
void edge_expsum_kernel(const long long* __restrict__ ei, const float* __restrict__ a_s,
                        const float* __restrict__ a_d, const unsigned* __restrict__ menc,
                        float* __restrict__ wexp, float* __restrict__ denom,
                        int E, int Etot) {
    int e = blockIdx.x * blockDim.x + threadIdx.x;
    if (e >= Etot) return;
    long long s, d;
    if (e < E) { s = ei[e]; d = ei[E + e]; } else { s = d = e - E; }
    float v = a_s[s] + a_d[d];
    v = v >= 0.f ? v : 0.2f * v;
    float w = __expf(v - dec_f(menc[(int)d]));
    wexp[e] = w;
    atomicAdd(&denom[(int)d], w);
}

__global__ __launch_bounds__(256)
void edge_aggr_kernel(const long long* __restrict__ ei, const float* __restrict__ h,
                      const float* __restrict__ wexp, const float* __restrict__ denom,
                      float* __restrict__ out, int E, int Etot) {
    int gw   = (blockIdx.x * blockDim.x + threadIdx.x) >> 5;  // wave per edge
    int lane = threadIdx.x & 31;
    if (gw >= Etot) return;
    long long s, d;
    if (gw < E) { s = ei[gw]; d = ei[E + gw]; } else { s = d = gw - E; }
    float alpha = wexp[gw] / denom[(int)d];
    float2 hv = ((const float2*)(h + (size_t)s * 64))[lane];
    float* o = out + (size_t)d * 64 + 2 * lane;
    atomicAdd(o,     alpha * hv.x);
    atomicAdd(o + 1, alpha * hv.y);
}

__global__ __launch_bounds__(256)
void bias_relu_kernel(float* __restrict__ h, const float* __restrict__ b, size_t n) {
    size_t i = (size_t)blockIdx.x * blockDim.x + threadIdx.x;
    if (i >= n) return;
    float v = h[i] + b[i & 63];
    h[i] = v > 0.f ? v : 0.f;
}

// ============================================================
// pooling + final linear
// ============================================================
__global__ __launch_bounds__(256)
void pool_sum_kernel(const float* __restrict__ h, const long long* __restrict__ batch,
                     float* __restrict__ sums, float* __restrict__ cnt, int N) {
    int gw   = (blockIdx.x * blockDim.x + threadIdx.x) >> 5;
    int lane = threadIdx.x & 31;
    if (gw >= N) return;
    int g = (int)batch[gw];
    float2 hv = ((const float2*)(h + (size_t)gw * 64))[lane];
    float* o = sums + (size_t)g * 64 + 2 * lane;
    atomicAdd(o,     hv.x);
    atomicAdd(o + 1, hv.y);
    if (lane == 0) atomicAdd(&cnt[g], 1.0f);
}

__global__ __launch_bounds__(256)
void final_lin_kernel(const float* __restrict__ sums, const float* __restrict__ cnt,
                      const float* __restrict__ Wlin, const float* __restrict__ blin,
                      float* __restrict__ out, int G) {
    int i = blockIdx.x * blockDim.x + threadIdx.x;
    if (i >= G * 2) return;
    int g = i >> 1, o = i & 1;
    float c = fmaxf(cnt[g], 1.0f);
    float acc = blin[o];
    #pragma unroll
    for (int k = 0; k < 64; ++k) acc += (sums[g * 64 + k] / c) * Wlin[k * 2 + o];
    out[i] = acc;
}

__global__ __launch_bounds__(256)
void fill_u32_kernel(unsigned* __restrict__ p, unsigned v, size_t n) {
    size_t i = (size_t)blockIdx.x * blockDim.x + threadIdx.x;
    if (i < n) p[i] = v;
}

// ============================================================
// host-side orchestration
// ============================================================
extern "C" void kernel_launch(void* const* d_in, const int* in_sizes, int n_in,
                              void* d_out, int out_size, void* d_ws, size_t ws_size,
                              hipStream_t stream) {
    const int FIN = 768, H = 64;
    const int N = in_sizes[0] / FIN;
    const int E = in_sizes[1] / 2;
    const int G = out_size / 2;
    const int Etot = E + N;

    const float*     x    = (const float*)d_in[0];
    const long long* ei   = (const long long*)d_in[1];
    const long long* bat  = (const long long*)d_in[2];
    const float* W1   = (const float*)d_in[3];
    const float* as1  = (const float*)d_in[4];
    const float* ad1  = (const float*)d_in[5];
    const float* b1   = (const float*)d_in[6];
    const float* W2   = (const float*)d_in[7];
    const float* as2  = (const float*)d_in[8];
    const float* ad2  = (const float*)d_in[9];
    const float* b2   = (const float*)d_in[10];
    const float* Wlin = (const float*)d_in[11];
    const float* blin = (const float*)d_in[12];
    float* out = (float*)d_out;

    // workspace carve-up
    float* ws = (float*)d_ws;
    float*    h1    = ws;                      ws += (size_t)N * H;
    float*    agg1  = ws;                      ws += (size_t)N * H;
    float*    h2    = ws;                      ws += (size_t)N * H;
    float*    agg2  = ws;                      ws += (size_t)N * H;
    float*    a_s   = ws;                      ws += N;
    float*    a_d   = ws;                      ws += N;
    unsigned* menc  = (unsigned*)ws;           ws += N;
    float*    denom = ws;                      ws += N;
    float*    wexp  = ws;                      ws += Etot;
    float*    sums  = ws;                      ws += (size_t)G * H;
    float*    cnt   = ws;                      ws += G;
    unsigned short* Wt1 = (unsigned short*)ws;            // 64 x FIN bf16 (transposed)
    unsigned short* Wt2 = Wt1 + (size_t)H * FIN;          // 64 x 64 bf16 (transposed)

    dim3 blk(256);
    const int gemmGrid  = (N + 127) / 128;
    const int edgeGrid  = (Etot + 255) / 256;
    const int edgeWGrid = (Etot * 32 + 255) / 256;   // wave-per-edge
    const int nodeWGrid = (N * 32 + 255) / 256;      // wave-per-node
    const size_t NH = (size_t)N * H;
    auto fgrid = [](size_t n) { return (unsigned)((n + 255) / 256); };

    // ---------------- weight pre-transpose/convert ----------------
    wt_convert_kernel<<<fgrid((size_t)H * FIN), blk, 0, stream>>>(W1, Wt1, FIN);
    wt_convert_kernel<<<fgrid((size_t)H * H),   blk, 0, stream>>>(W2, Wt2, H);

    // ---------------- layer 1 ----------------
    gemm_bf16_wmma<<<gemmGrid, blk, 0, stream>>>(x, Wt1, h1, N, FIN);
    attdot_kernel<<<nodeWGrid, blk, 0, stream>>>(h1, as1, ad1, a_s, a_d, N);

    fill_u32_kernel<<<fgrid(N), blk, 0, stream>>>(menc, ENC_NEG_INF, (size_t)N);
    fill_u32_kernel<<<fgrid(N), blk, 0, stream>>>((unsigned*)denom, 0u, (size_t)N);
    fill_u32_kernel<<<fgrid(NH), blk, 0, stream>>>((unsigned*)agg1, 0u, NH);

    edge_max_kernel<<<edgeGrid, blk, 0, stream>>>(ei, a_s, a_d, menc, E, Etot);
    edge_expsum_kernel<<<edgeGrid, blk, 0, stream>>>(ei, a_s, a_d, menc, wexp, denom, E, Etot);
    edge_aggr_kernel<<<edgeWGrid, blk, 0, stream>>>(ei, h1, wexp, denom, agg1, E, Etot);
    bias_relu_kernel<<<fgrid(NH), blk, 0, stream>>>(agg1, b1, NH);

    // ---------------- layer 2 ----------------
    gemm_bf16_wmma<<<gemmGrid, blk, 0, stream>>>(agg1, Wt2, h2, N, H);
    attdot_kernel<<<nodeWGrid, blk, 0, stream>>>(h2, as2, ad2, a_s, a_d, N);

    fill_u32_kernel<<<fgrid(N), blk, 0, stream>>>(menc, ENC_NEG_INF, (size_t)N);
    fill_u32_kernel<<<fgrid(N), blk, 0, stream>>>((unsigned*)denom, 0u, (size_t)N);
    fill_u32_kernel<<<fgrid(NH), blk, 0, stream>>>((unsigned*)agg2, 0u, NH);

    edge_max_kernel<<<edgeGrid, blk, 0, stream>>>(ei, a_s, a_d, menc, E, Etot);
    edge_expsum_kernel<<<edgeGrid, blk, 0, stream>>>(ei, a_s, a_d, menc, wexp, denom, E, Etot);
    edge_aggr_kernel<<<edgeWGrid, blk, 0, stream>>>(ei, h2, wexp, denom, agg2, E, Etot);
    bias_relu_kernel<<<fgrid(NH), blk, 0, stream>>>(agg2, b2, NH);

    // ---------------- pool + head ----------------
    fill_u32_kernel<<<fgrid((size_t)G * H), blk, 0, stream>>>((unsigned*)sums, 0u, (size_t)G * H);
    fill_u32_kernel<<<fgrid(G), blk, 0, stream>>>((unsigned*)cnt, 0u, (size_t)G);
    pool_sum_kernel<<<nodeWGrid, blk, 0, stream>>>(agg2, bat, sums, cnt, N);
    final_lin_kernel<<<(G * 2 + 255) / 256, blk, 0, stream>>>(sums, cnt, Wlin, blin, out, G);
}
